// FPN2MLPFeatureExtractorAffinitySpatial_44126493999175
// MI455X (gfx1250) — compile-verified, matrix operand order checked by reading, not run on hardware
//
#include <hip/hip_runtime.h>
#include <hip/hip_bf16.h>

// ---------------- CDNA5 WMMA helpers (wave32, 16x16x32 bf16) ----------------
typedef __attribute__((ext_vector_type(16))) __bf16 bf16x16;
typedef __attribute__((ext_vector_type(8)))  float  f32x8;

union FragU { bf16x16 v; uint4 u[2]; };

// Load a 16x32 A-fragment from row-major A[rows, ld] (rows = M), or a 32x16
// B-fragment from "N-major" storage S where B[k,n] = S[n*ld + k].
// ISA layout: lanes 0-15 hold (row = lane, K in {0..7, 16..23});
//             lanes 16-31 hold (row = lane-16, K in {8..15, 24..31}).
static __device__ __forceinline__ bf16x16
load_frag16(const __bf16* __restrict__ p, long row0, long ld, long k0) {
  const int lane = threadIdx.x & 31;
  const long r  = row0 + (lane & 15);
  const long kb = k0 + ((lane >> 4) << 3);
  const __bf16* q = p + r * ld + kb;
  FragU f;
  f.u[0] = *(const uint4*)(q);
  f.u[1] = *(const uint4*)(q + 16);
  return f.v;
}

static __device__ __forceinline__ f32x8
wmma_bf16(bf16x16 a, bf16x16 b, f32x8 c) {
  return __builtin_amdgcn_wmma_f32_16x16x32_bf16(
      /*neg_a=*/false, a, /*neg_b=*/false, b,
      /*c_mod=*/(short)0, c, /*reuse_a=*/false, /*reuse_b=*/false);
}

// ---------------- problem constants ----------------
#define QROWS 25088   // B*H*W
#define SROWS 3136    // NS*HS*WS
#define CIN   256
#define CKC   128
#define HW_Q  49
#define FIN   12544   // 2*CKC*HW_Q
#define REPN  1024
#define BATCH 512

// ---------------- pack / convert kernels ----------------
// [N, C, HW] f32 -> [N*HW, C] bf16  (pixel-major for A fragments)
__global__ void k_pack_nchw(const float* __restrict__ in, __bf16* __restrict__ out,
                            int C, int HW, long total) {
  long idx = (long)blockIdx.x * blockDim.x + threadIdx.x;
  if (idx >= total) return;
  long p = idx / C;
  int  c = (int)(idx - p * C);
  long n = p / HW;
  int hw = (int)(p - n * HW);
  out[idx] = (__bf16)in[(n * C + c) * HW + hw];
}

__global__ void k_cvt_bf16(const float* __restrict__ in, __bf16* __restrict__ out,
                           long n) {
  long idx = (long)blockIdx.x * blockDim.x + threadIdx.x;
  if (idx < n) out[idx] = (__bf16)in[idx];
}

// ---------------- projection GEMM: P = A[rows,256] @ W^T + bias ----------------
// mode 0: out row-major [rows,128]          (qk, sk)
// mode 1: out scattered into fuse at channel c (qv, channels 0..127)
// mode 2: out transposed [128, SROWS]       (svT)
__global__ __launch_bounds__(128)
void k_proj(const __bf16* __restrict__ A, const __bf16* __restrict__ W,
            const float* __restrict__ bias, __bf16* __restrict__ out,
            int rows, int mode) {
  const int wave = (int)((blockIdx.x * blockDim.x + threadIdx.x) >> 5);
  const int lane = threadIdx.x & 31;
  const int tile_m = wave >> 3;        // 8 N-tiles of 16 cover 128 outputs
  const int tile_n = wave & 7;
  const int m0 = tile_m * 16, n0 = tile_n * 16;
  if (m0 >= rows) return;

  f32x8 acc = {};
#pragma unroll
  for (int k0 = 0; k0 < CIN; k0 += 32) {
    bf16x16 a = load_frag16(A, m0, CIN, k0);
    bf16x16 b = load_frag16(W, n0, CIN, k0);   // B[k,n] = W[n*256 + k]
    acc = wmma_bf16(a, b, acc);
  }

  const int grp = lane >> 4, nloc = lane & 15;
  const int c = n0 + nloc;
  const float bv = bias[c];
#pragma unroll
  for (int r = 0; r < 8; ++r) {
    const int m = m0 + r + 8 * grp;
    const __bf16 o = (__bf16)(acc[r] + bv);
    if (mode == 0) {
      out[(long)m * CKC + c] = o;
    } else if (mode == 1) {            // qv -> fuse[b, c*49 + hw]
      const int b = m / HW_Q, hw = m - b * HW_Q;
      out[(long)b * FIN + c * HW_Q + hw] = o;
    } else {                           // svT[c, m]
      out[(long)c * SROWS + m] = o;
    }
  }
}

// ---------------- flash attention: softmax(qk @ sk^T) @ sv -> fuse ----------------
__global__ __launch_bounds__(256)
void k_attn(const __bf16* __restrict__ qk, const __bf16* __restrict__ sk,
            const __bf16* __restrict__ svT, __bf16* __restrict__ fuse) {
  __shared__ __align__(16) __bf16 lds[8 * 16 * 32];   // per-wave 16x32 scratch
  const int wv   = threadIdx.x >> 5;
  const int lane = threadIdx.x & 31;
  const int tile = blockIdx.x * 8 + wv;               // 1568 query tiles
  const int m0 = tile * 16;
  __bf16* scr = lds + wv * (16 * 32);
  const int grp = lane >> 4, nloc = lane & 15;

  // Query fragments held in registers for the whole pass (K=128 -> 4 chunks).
  bf16x16 aq[4];
#pragma unroll
  for (int kc = 0; kc < 4; ++kc) aq[kc] = load_frag16(qk, m0, CKC, kc * 32);

  f32x8 acc[8] = {};                                  // 16 x 128 f32 accumulator
  float mrow[8], lrow[8];
#pragma unroll
  for (int r = 0; r < 8; ++r) { mrow[r] = -__builtin_inff(); lrow[r] = 0.f; }

  for (int s0 = 0; s0 < SROWS; s0 += 32) {
    // scores: two 16x16 tiles over this 32-support chunk
    f32x8 sA = {}, sB = {};
#pragma unroll
    for (int kc = 0; kc < 4; ++kc) {
      bf16x16 b0 = load_frag16(sk, s0,      CKC, kc * 32);
      bf16x16 b1 = load_frag16(sk, s0 + 16, CKC, kc * 32);
      sA = wmma_bf16(aq[kc], b0, sA);
      sB = wmma_bf16(aq[kc], b1, sB);
    }

    // online softmax (rows live in 16-lane groups; reduce with shfl_xor)
#pragma unroll
    for (int r = 0; r < 8; ++r) {
      float v0 = sA[r], v1 = sB[r];
      float mx = fmaxf(v0, v1);
#pragma unroll
      for (int off = 1; off < 16; off <<= 1)
        mx = fmaxf(mx, __shfl_xor(mx, off, 32));
      const float mnew  = fmaxf(mrow[r], mx);
      const float scale = __expf(mrow[r] - mnew);
      const float p0 = __expf(v0 - mnew);
      const float p1 = __expf(v1 - mnew);
      float ps = p0 + p1;
#pragma unroll
      for (int off = 1; off < 16; off <<= 1)
        ps += __shfl_xor(ps, off, 32);
      lrow[r] = lrow[r] * scale + ps;
      mrow[r] = mnew;
#pragma unroll
      for (int nt = 0; nt < 8; ++nt) acc[nt][r] *= scale;
      const int m = r + 8 * grp;                       // C-layout row
      scr[m * 32 + nloc]      = (__bf16)p0;
      scr[m * 32 + nloc + 16] = (__bf16)p1;
    }

    // reload probabilities as a 16x32 A-fragment (same-wave DS ops are in-order)
    bf16x16 pfrag;
    {
      const __bf16* q = scr + (lane & 15) * 32 + ((lane >> 4) << 3);
      FragU f;
      f.u[0] = *(const uint4*)(q);
      f.u[1] = *(const uint4*)(q + 16);
      pfrag = f.v;
    }

    // acc += P @ sv   (B[k,n] = svT[(c)*SROWS + s0+k], contiguous per lane)
#pragma unroll
    for (int nt = 0; nt < 8; ++nt) {
      bf16x16 bsv = load_frag16(svT, nt * 16, SROWS, s0);
      acc[nt] = wmma_bf16(pfrag, bsv, acc[nt]);
    }
  }

  // normalize and scatter into fuse channels 128..255
#pragma unroll
  for (int r = 0; r < 8; ++r) {
    const int m = m0 + r + 8 * grp;
    const int b = m / HW_Q, hw = m - b * HW_Q;
    const float inv = 1.f / lrow[r];
#pragma unroll
    for (int nt = 0; nt < 8; ++nt) {
      const int c = CKC + nt * 16 + nloc;
      fuse[(long)b * FIN + c * HW_Q + hw] = (__bf16)(acc[nt][r] * inv);
    }
  }
}

// ---------------- fully-connected GEMM: Y = relu(A @ W^T + b) ----------------
// block = 8 waves, 128(M) x 64(N) tile; wave w owns rows [bm*128+w*16, +16)
__global__ __launch_bounds__(256)
void k_fc(const __bf16* __restrict__ A, const __bf16* __restrict__ W,
          const float* __restrict__ bias, void* __restrict__ out,
          int M, int N, int K, int out_is_f32) {
  const int wave = threadIdx.x >> 5;
  const int lane = threadIdx.x & 31;
  const int m0 = blockIdx.x * 128 + wave * 16;
  const int n0 = blockIdx.y * 64;

  f32x8 acc[4] = {};
  for (int k0 = 0; k0 < K; k0 += 32) {
    if (k0 + 64 < K)   // pull next A chunk toward the WGP while WMMAs run
      __builtin_prefetch((const void*)(A + (long)(m0 + (lane & 15)) * K + k0 + 64), 0, 0);
    bf16x16 a = load_frag16(A, m0, K, k0);
#pragma unroll
    for (int nt = 0; nt < 4; ++nt) {
      bf16x16 b = load_frag16(W, n0 + nt * 16, K, k0);
      acc[nt] = wmma_bf16(a, b, acc[nt]);
    }
  }

  const int grp = lane >> 4, nloc = lane & 15;
#pragma unroll
  for (int nt = 0; nt < 4; ++nt) {
    const int n = n0 + nt * 16 + nloc;
    const float bv = bias[n];
#pragma unroll
    for (int r = 0; r < 8; ++r) {
      const int m = m0 + r + 8 * grp;
      const float v = fmaxf(acc[nt][r] + bv, 0.f);
      if (out_is_f32) ((float*)out)[(long)m * N + n] = v;
      else            ((__bf16*)out)[(long)m * N + n] = (__bf16)v;
    }
  }
}

// ---------------- host orchestration ----------------
extern "C" void kernel_launch(void* const* d_in, const int* in_sizes, int n_in,
                              void* d_out, int out_size, void* d_ws, size_t ws_size,
                              hipStream_t stream) {
  const float* x   = (const float*)d_in[0];
  const float* sup = (const float*)d_in[1];
  const float* Wqv = (const float*)d_in[2];  const float* bqv = (const float*)d_in[3];
  const float* Wqk = (const float*)d_in[4];  const float* bqk = (const float*)d_in[5];
  const float* Wsv = (const float*)d_in[6];  const float* bsv = (const float*)d_in[7];
  const float* Wsk = (const float*)d_in[8];  const float* bsk = (const float*)d_in[9];
  const float* W6c = (const float*)d_in[10]; const float* b6c = (const float*)d_in[11];
  const float* W7c = (const float*)d_in[12]; const float* b7c = (const float*)d_in[13];
  const float* W6r = (const float*)d_in[14]; const float* b6r = (const float*)d_in[15];
  const float* W7r = (const float*)d_in[16]; const float* b7r = (const float*)d_in[17];

  char* ws = (char*)d_ws;
  auto alloc = [&](size_t bytes) -> char* {
    char* p = ws; ws += (bytes + 255) & ~(size_t)255; return p;
  };
  __bf16* xp   = (__bf16*)alloc((size_t)QROWS * CIN * 2);
  __bf16* sp   = (__bf16*)alloc((size_t)SROWS * CIN * 2);
  __bf16* wqv  = (__bf16*)alloc((size_t)CKC * CIN * 2);
  __bf16* wqk  = (__bf16*)alloc((size_t)CKC * CIN * 2);
  __bf16* wsv  = (__bf16*)alloc((size_t)CKC * CIN * 2);
  __bf16* wsk  = (__bf16*)alloc((size_t)CKC * CIN * 2);
  __bf16* qkb  = (__bf16*)alloc((size_t)QROWS * CKC * 2);
  __bf16* skb  = (__bf16*)alloc((size_t)SROWS * CKC * 2);
  __bf16* svT  = (__bf16*)alloc((size_t)CKC * SROWS * 2);
  __bf16* fuse = (__bf16*)alloc((size_t)BATCH * FIN * 2);
  __bf16* w6cb = (__bf16*)alloc((size_t)REPN * FIN * 2);
  __bf16* w6rb = (__bf16*)alloc((size_t)REPN * FIN * 2);
  __bf16* w7cb = (__bf16*)alloc((size_t)REPN * REPN * 2);
  __bf16* w7rb = (__bf16*)alloc((size_t)REPN * REPN * 2);
  __bf16* xc6  = (__bf16*)alloc((size_t)BATCH * REPN * 2);
  __bf16* xr6  = (__bf16*)alloc((size_t)BATCH * REPN * 2);

  const int T = 256;
  auto nblk = [](long n, int t) { return (unsigned)((n + t - 1) / t); };

  // 1) precision packing (memory-bound; one pass over mandatory traffic)
  k_pack_nchw<<<nblk((long)QROWS * CIN, T), T, 0, stream>>>(x,   xp, CIN, HW_Q,   (long)QROWS * CIN);
  k_pack_nchw<<<nblk((long)SROWS * CIN, T), T, 0, stream>>>(sup, sp, CIN, 14 * 14, (long)SROWS * CIN);
  k_cvt_bf16<<<nblk((long)CKC * CIN, T), T, 0, stream>>>(Wqv, wqv, (long)CKC * CIN);
  k_cvt_bf16<<<nblk((long)CKC * CIN, T), T, 0, stream>>>(Wqk, wqk, (long)CKC * CIN);
  k_cvt_bf16<<<nblk((long)CKC * CIN, T), T, 0, stream>>>(Wsv, wsv, (long)CKC * CIN);
  k_cvt_bf16<<<nblk((long)CKC * CIN, T), T, 0, stream>>>(Wsk, wsk, (long)CKC * CIN);
  k_cvt_bf16<<<nblk((long)REPN * FIN, T), T, 0, stream>>>(W6c, w6cb, (long)REPN * FIN);
  k_cvt_bf16<<<nblk((long)REPN * FIN, T), T, 0, stream>>>(W6r, w6rb, (long)REPN * FIN);
  k_cvt_bf16<<<nblk((long)REPN * REPN, T), T, 0, stream>>>(W7c, w7cb, (long)REPN * REPN);
  k_cvt_bf16<<<nblk((long)REPN * REPN, T), T, 0, stream>>>(W7r, w7rb, (long)REPN * REPN);

  // 2) 1x1-conv projections (WMMA)
  {
    unsigned wavesQ = (QROWS / 16) * 8, wavesS = (SROWS / 16) * 8;
    k_proj<<<wavesQ * 32 / 128, 128, 0, stream>>>(xp, wqk, bqk, qkb,  QROWS, 0);
    k_proj<<<wavesQ * 32 / 128, 128, 0, stream>>>(xp, wqv, bqv, fuse, QROWS, 1);
    k_proj<<<wavesS * 32 / 128, 128, 0, stream>>>(sp, wsk, bsk, skb,  SROWS, 0);
    k_proj<<<wavesS * 32 / 128, 128, 0, stream>>>(sp, wsv, bsv, svT,  SROWS, 2);
  }

  // 3) fused attention (flash-style, sim never materialized)
  k_attn<<<(QROWS / 16) / 8, 256, 0, stream>>>(qkb, skb, svT, fuse);

  // 4) MLP heads; fc7 writes f32 directly into d_out (xc first, then xr)
  dim3 g6(BATCH / 128, REPN / 64), g7(BATCH / 128, REPN / 64);
  k_fc<<<g6, 256, 0, stream>>>(fuse, w6cb, b6c, xc6, BATCH, REPN, FIN, 0);
  k_fc<<<g6, 256, 0, stream>>>(fuse, w6rb, b6r, xr6, BATCH, REPN, FIN, 0);
  float* out_c = (float*)d_out;
  float* out_r = out_c + (size_t)BATCH * REPN;
  k_fc<<<g7, 256, 0, stream>>>(xc6, w7cb, b7c, out_c, BATCH, REPN, REPN, 1);
  k_fc<<<g7, 256, 0, stream>>>(xr6, w7rb, b7r, out_r, BATCH, REPN, REPN, 1);
}
